// Network_63969242907101
// MI455X (gfx1250) — matrix operand order repaired and checked by reading, not compiled, not run on hardware
//
#include <hip/hip_runtime.h>
#include <hip/hip_bf16.h>
#include <math.h>

typedef _Float16 f16;
typedef _Float16 v16h __attribute__((ext_vector_type(16)));
typedef _Float16 v8h  __attribute__((ext_vector_type(8)));
typedef float    v8f  __attribute__((ext_vector_type(8)));

#define NATOM 32768
#define KNBR  16
#define EDGES (NATOM * KNBR)
#define MAXRC 2.0f

// ---------------------------------------------------------------------------
// Weight prep: src f32 [K][N] row-major  ->  dst f16 [Npad][Kpad] (transposed,
// zero padded) so a WMMA B-fragment lane reads 16 contiguous f16 (32 bytes).
// ---------------------------------------------------------------------------
__global__ void pad_weights(const float* __restrict__ src, f16* __restrict__ dst,
                            int K, int Nn, int Kpad, int Npad) {
    int t = blockIdx.x * blockDim.x + threadIdx.x;
    int total = Npad * Kpad;
    if (t >= total) return;
    int n = t / Kpad;
    int k = t % Kpad;
    float v = (n < Nn && k < K) ? src[k * Nn + n] : 0.f;
    dst[t] = (f16)v;
}

// ---------------------------------------------------------------------------
// Per-edge geometry: rvec, r, spherical harmonics l=1..3 (15 vals), cosine^2
// radial basis (10 -> padded 32 f16), cutoff mask.
// ---------------------------------------------------------------------------
__global__ void geom_kernel(const float* __restrict__ coords,
                            const float* __restrict__ nbr_shift,
                            const int*   __restrict__ nbr_idx,
                            f16*   __restrict__ basis,
                            float* __restrict__ Yv,
                            float* __restrict__ maskv) {
    int e = blockIdx.x * blockDim.x + threadIdx.x;
    if (e >= EDGES) return;
    int i = e >> 4;
    int j = nbr_idx[e];
    float dx = coords[(size_t)j * 3 + 0] + nbr_shift[(size_t)e * 3 + 0] - coords[(size_t)i * 3 + 0];
    float dy = coords[(size_t)j * 3 + 1] + nbr_shift[(size_t)e * 3 + 1] - coords[(size_t)i * 3 + 1];
    float dz = coords[(size_t)j * 3 + 2] + nbr_shift[(size_t)e * 3 + 2] - coords[(size_t)i * 3 + 2];
    float r  = sqrtf(dx * dx + dy * dy + dz * dz);
    float rs = fmaxf(r, 1e-6f);
    float x = dx / rs, y = dy / rs, z = dz / rs;
    float x2 = x * x, y2 = y * y, z2 = z * z;

    float* Ye = Yv + (size_t)e * 16;
    Ye[0]  = 0.4886025f * y;
    Ye[1]  = 0.4886025f * z;
    Ye[2]  = 0.4886025f * x;
    Ye[3]  = 1.0925484f * x * y;
    Ye[4]  = 1.0925484f * y * z;
    Ye[5]  = 0.3153916f * (3.f * z2 - 1.f);
    Ye[6]  = 1.0925484f * x * z;
    Ye[7]  = 0.5462742f * (x2 - y2);
    Ye[8]  = 0.5900436f * y * (3.f * x2 - y2);
    Ye[9]  = 2.8906114f * x * y * z;
    Ye[10] = 0.4570458f * y * (5.f * z2 - 1.f);
    Ye[11] = 0.3731763f * z * (5.f * z2 - 3.f);
    Ye[12] = 0.4570458f * x * (5.f * z2 - 1.f);
    Ye[13] = 1.4453057f * (x2 - y2) * z;
    Ye[14] = 0.5900436f * x * (x2 - 3.f * y2);
    Ye[15] = 0.f;

    f16* Be = basis + (size_t)e * 32;
    const float step = MAXRC / 9.f;
#pragma unroll
    for (int b = 0; b < 10; b++) {
        float xr = (r - step * (float)b) / step;
        float v = 0.f;
        if (fabsf(xr) < 1.f) {
            float c = __cosf(0.5f * 3.14159265358979f * xr);
            v = c * c;
        }
        Be[b] = (f16)v;
    }
#pragma unroll
    for (int b = 10; b < 32; b++) Be[b] = (f16)0.f;

    maskv[e] = (r < MAXRC && r > 1e-6f) ? 1.f : 0.f;
}

// ---------------------------------------------------------------------------
// Fused radial MLP:  w = (relu(relu(B@W1)@W2)@W3) * mask
// 128 edges per block, 4 waves, WMMA f32_16x16x32_f16, hidden acts in LDS.
// Each wave owns n-strips: B (weight) fragments are hoisted into registers
// once per strip and reused across all 8 m-tiles (8x less B traffic, no
// per-tile load-wait before the WMMAs).
// Fragment layouts: A 16x32 / B 32x16 in contiguous k order (lane half h: k = 16h+s); f32 C per ISA.
// ---------------------------------------------------------------------------
__global__ __launch_bounds__(128) void radial_mlp(
    const f16* __restrict__ basis,
    const f16* __restrict__ W1t,   // [128][32]
    const f16* __restrict__ W2t,   // [128][128]
    const f16* __restrict__ W3t,   // [nwPad][128]
    const float* __restrict__ maskv,
    f16* __restrict__ wE,          // [EDGES][256]
    int nwPad) {
    __shared__ f16 H1[128][128];
    __shared__ f16 H2[128][128];

    const int tid  = threadIdx.x;
    const int wave = tid >> 5;
    const int lane = tid & 31;
    const int lh   = lane >> 4;   // K-half select
    const int ll   = lane & 15;   // row / column within tile
    const int e0   = blockIdx.x * 128;

    // ---- Stage 1: H1 = relu(basis @ W1), K = 32 (one WMMA step)
    for (int ns_ = wave; ns_ < 8; ns_ += 4) {
        int n0 = ns_ << 4;
        v16h b = *(const v16h*)(W1t + (size_t)(n0 + ll) * 32 + lh * 16);
#pragma unroll
        for (int mt = 0; mt < 8; mt++) {
            int m0 = mt << 4;
            v16h a;
            const f16* ap = basis + ((size_t)(e0 + m0 + ll)) * 32 + lh * 16;
            *((v8h*)&a)     = *(const v8h*)(ap);
            *((v8h*)&a + 1) = *(const v8h*)(ap + 8);
            v8f c = {};
            c = __builtin_amdgcn_wmma_f32_16x16x32_f16(false, a, false, b,
                                                       (short)0, c, false, false);
#pragma unroll
            for (int r = 0; r < 8; r++) {
                float v = c[r];
                v = v > 0.f ? v : 0.f;
                H1[m0 + r + lh * 8][n0 + ll] = (f16)v;
            }
        }
    }
    __syncthreads();

    // ---- Stage 2: H2 = relu(H1 @ W2), K = 128 (4 WMMA steps, B resident)
    for (int ns_ = wave; ns_ < 8; ns_ += 4) {
        int n0 = ns_ << 4;
        v16h bk[4];
#pragma unroll
        for (int ks = 0; ks < 4; ks++)
            bk[ks] = *(const v16h*)(W2t + (size_t)(n0 + ll) * 128 + ks * 32 + lh * 16);
#pragma unroll
        for (int mt = 0; mt < 8; mt++) {
            int m0 = mt << 4;
            v8f c = {};
#pragma unroll
            for (int ks = 0; ks < 4; ks++) {
                int k0 = ks * 32;
                v16h a;
                const f16* ap = &H1[m0 + ll][k0 + lh * 16];
                *((v8h*)&a)     = *(const v8h*)(ap);
                *((v8h*)&a + 1) = *(const v8h*)(ap + 8);
                c = __builtin_amdgcn_wmma_f32_16x16x32_f16(false, a, false, bk[ks],
                                                           (short)0, c, false, false);
            }
#pragma unroll
            for (int r = 0; r < 8; r++) {
                float v = c[r];
                v = v > 0.f ? v : 0.f;
                H2[m0 + r + lh * 8][n0 + ll] = (f16)v;
            }
        }
    }
    __syncthreads();

    // ---- Stage 3: w = (H2 @ W3) * mask, write to global (stride 256)
    const int nt3 = nwPad >> 4;
    for (int ntile = wave; ntile < nt3; ntile += 4) {
        int n0 = ntile << 4;
        v16h bk[4];
#pragma unroll
        for (int ks = 0; ks < 4; ks++)
            bk[ks] = *(const v16h*)(W3t + (size_t)(n0 + ll) * 128 + ks * 32 + lh * 16);
#pragma unroll
        for (int mt = 0; mt < 8; mt++) {
            int m0 = mt << 4;
            v8f c = {};
#pragma unroll
            for (int ks = 0; ks < 4; ks++) {
                int k0 = ks * 32;
                v16h a;
                const f16* ap = &H2[m0 + ll][k0 + lh * 16];
                *((v8h*)&a)     = *(const v8h*)(ap);
                *((v8h*)&a + 1) = *(const v8h*)(ap + 8);
                c = __builtin_amdgcn_wmma_f32_16x16x32_f16(false, a, false, bk[ks],
                                                           (short)0, c, false, false);
            }
#pragma unroll
            for (int r = 0; r < 8; r++) {
                int row  = m0 + r + lh * 8;
                size_t e = (size_t)e0 + row;
                float v  = c[r] * maskv[e];
                wE[e * 256 + n0 + ll] = (f16)v;
            }
        }
    }
}

// ---------------------------------------------------------------------------
// Equivariant conv + gating, templated on layer index so all loops fully
// unroll and the contiguous f16 weight / f32 feature reads coalesce into
// wide vector loads (bandwidth-bound kernel). One thread per atom.
// Feature layout per atom: [ l0:4 | l1:4x3 | l2:4x5 | l3:4x7 ] = 64 f32.
// ---------------------------------------------------------------------------
template <int LI>
__global__ void conv_layer(const float* __restrict__ fin,
                           const f16*   __restrict__ wE,
                           const float* __restrict__ Yv,
                           const int*   __restrict__ nbr_idx,
                           float* __restrict__ fout) {
    int n = blockIdx.x * blockDim.x + threadIdx.x;
    if (n >= NATOM) return;

    constexpr int  I0   = (LI == 0) ? 1 : 4;
    constexpr int  O0   = (LI == 3) ? 64 : 16;
    constexpr bool GATE = (LI != 3);

    float out0[O0];
#pragma unroll
    for (int o = 0; o < O0; o++) out0[o] = 0.f;
    float ns[4][15];  // l=1 -> [0..2], l=2 -> [3..7], l=3 -> [8..14]
#pragma unroll
    for (int o = 0; o < 4; o++)
#pragma unroll
        for (int m = 0; m < 15; m++) ns[o][m] = 0.f;

    for (int k = 0; k < KNBR; k++) {
        size_t e = (size_t)n * KNBR + k;
        int j = nbr_idx[e];
        const f16*   w  = wE + e * 256;
        const float* Ye = Yv + e * 16;

        float f0[I0];
        if (LI == 0) {
            f0[0] = 0.5f;  // ones / 2
        } else {
#pragma unroll
            for (int i = 0; i < I0; i++) f0[i] = 0.5f * fin[(size_t)j * 64 + i];
        }

        int idx = 0;
        // scalar output path: wA0 [O0][I0]
#pragma unroll
        for (int o = 0; o < O0; o++) {
            float a = 0.f;
#pragma unroll
            for (int i = 0; i < I0; i++) a += (float)w[idx++] * f0[i];
            out0[o] += a;
        }
        if (GATE) {
            int yoff = 0;
#pragma unroll
            for (int l = 1; l <= 3; l++) {
                const int dim = 2 * l + 1;
                // Y-injection path: wB [4][I0] * f0 * Y_l
#pragma unroll
                for (int o = 0; o < 4; o++) {
                    float cY = 0.f;
#pragma unroll
                    for (int i = 0; i < I0; i++) cY += (float)w[idx++] * f0[i];
#pragma unroll
                    for (int m = 0; m < dim; m++) ns[o][yoff + m] += cY * Ye[yoff + m];
                }
                // l-diagonal path: wA [4][4] * fj[l]
                if (LI != 0) {
                    const int foff = (l == 1) ? 4 : (l == 2 ? 16 : 36);
#pragma unroll
                    for (int o = 0; o < 4; o++)
#pragma unroll
                        for (int i = 0; i < 4; i++) {
                            float wv = (float)w[idx++];
#pragma unroll
                            for (int m = 0; m < dim; m++)
                                ns[o][yoff + m] +=
                                    wv * 0.5f * fin[(size_t)j * 64 + foff + i * dim + m];
                        }
                }
                yoff += dim;
            }
        }
    }

    float* fo = fout + (size_t)n * 64;
    if (GATE) {
#pragma unroll
        for (int o = 0; o < 4; o++) fo[o] = fmaxf(out0[o], 0.f);
        float g[12];
#pragma unroll
        for (int t = 0; t < 12; t++) g[t] = 1.f / (1.f + __expf(-out0[4 + t]));
        int yoff = 0;
#pragma unroll
        for (int l = 1; l <= 3; l++) {
            const int dim  = 2 * l + 1;
            const int foff = (l == 1) ? 4 : (l == 2 ? 16 : 36);
#pragma unroll
            for (int o = 0; o < 4; o++)
#pragma unroll
                for (int m = 0; m < dim; m++)
                    fo[foff + o * dim + m] = ns[o][yoff + m] * g[4 * (l - 1) + o];
            yoff += dim;
        }
    } else {
#pragma unroll
        for (int o = 0; o < O0; o++) fo[o] = fmaxf(out0[o], 0.f);
    }
}

// ---------------------------------------------------------------------------
// AvgSpacial pooling: one block per channel, deterministic tree reduce.
// ---------------------------------------------------------------------------
__global__ void pool_kernel(const float* __restrict__ feat, float* __restrict__ pooled) {
    __shared__ float red[256];
    int c = blockIdx.x;
    float s = 0.f;
    for (int nn = threadIdx.x; nn < NATOM; nn += 256) s += feat[(size_t)nn * 64 + c];
    red[threadIdx.x] = s;
    __syncthreads();
    for (int off = 128; off > 0; off >>= 1) {
        if (threadIdx.x < off) red[threadIdx.x] += red[threadIdx.x + off];
        __syncthreads();
    }
    if (threadIdx.x == 0) pooled[c] = red[0] / (float)NATOM;
}

__global__ void head_kernel(const float* __restrict__ pooled,
                            const float* __restrict__ lin_w,
                            const float* __restrict__ lin_b,
                            float* __restrict__ out) {
    int t = threadIdx.x;
    if (t < 10) {
        float s = lin_b[t];
        for (int c = 0; c < 64; c++) s += pooled[c] * lin_w[c * 10 + t];
        out[t] = s;
    }
}

// ---------------------------------------------------------------------------
extern "C" void kernel_launch(void* const* d_in, const int* in_sizes, int n_in,
                              void* d_out, int out_size, void* d_ws, size_t ws_size,
                              hipStream_t stream) {
    (void)in_sizes; (void)n_in; (void)out_size; (void)ws_size;
    // setup_inputs() dict order:
    // 0 coords, 1 nbr_shift, 2 nbr_idx, 3..14 (w1_i,w2_i,w3_i)*4, 15 lin_w, 16 lin_b
    const float* coords    = (const float*)d_in[0];
    const float* nbr_shift = (const float*)d_in[1];
    const int*   nbr_idx   = (const int*)d_in[2];
    const float* w1[4]; const float* w2[4]; const float* w3[4];
    for (int i = 0; i < 4; i++) {
        w1[i] = (const float*)d_in[3 + 3 * i];
        w2[i] = (const float*)d_in[4 + 3 * i];
        w3[i] = (const float*)d_in[5 + 3 * i];
    }
    const float* lin_w = (const float*)d_in[15];
    const float* lin_b = (const float*)d_in[16];
    float* out = (float*)d_out;

    // workspace carve
    char* p = (char*)d_ws;
    auto carve = [&](size_t bytes) -> char* {
        char* r = p;
        p += (bytes + 255) & ~(size_t)255;
        return r;
    };
    f16*   basis  = (f16*)carve((size_t)EDGES * 32 * sizeof(f16));
    float* Yv     = (float*)carve((size_t)EDGES * 16 * sizeof(float));
    float* maskv  = (float*)carve((size_t)EDGES * sizeof(float));
    f16*   wEbuf  = (f16*)carve((size_t)EDGES * 256 * sizeof(f16));
    float* featA  = (float*)carve((size_t)NATOM * 64 * sizeof(float));
    float* featB  = (float*)carve((size_t)NATOM * 64 * sizeof(float));
    float* pooled = (float*)carve(64 * sizeof(float));

    const int nw[4]    = {28, 160, 160, 256};
    const int nwPad[4] = {32, 160, 160, 256};
    f16* W1t[4]; f16* W2t[4]; f16* W3t[4];
    for (int i = 0; i < 4; i++) {
        W1t[i] = (f16*)carve(128 * 32 * sizeof(f16));
        W2t[i] = (f16*)carve(128 * 128 * sizeof(f16));
        W3t[i] = (f16*)carve((size_t)nwPad[i] * 128 * sizeof(f16));
    }

    // 1) weight prep (transpose + pad to f16)
    for (int i = 0; i < 4; i++) {
        int tot;
        tot = 128 * 32;
        pad_weights<<<(tot + 255) / 256, 256, 0, stream>>>(w1[i], W1t[i], 10, 100, 32, 128);
        tot = 128 * 128;
        pad_weights<<<(tot + 255) / 256, 256, 0, stream>>>(w2[i], W2t[i], 100, 100, 128, 128);
        tot = nwPad[i] * 128;
        pad_weights<<<(tot + 255) / 256, 256, 0, stream>>>(w3[i], W3t[i], 100, nw[i], 128, nwPad[i]);
    }

    // 2) edge geometry
    geom_kernel<<<EDGES / 256, 256, 0, stream>>>(coords, nbr_shift, nbr_idx,
                                                 basis, Yv, maskv);

    // 3) four layers: WMMA radial MLP -> conv + gating
    // layer 0
    radial_mlp<<<EDGES / 128, 128, 0, stream>>>(basis, W1t[0], W2t[0], W3t[0],
                                                maskv, wEbuf, nwPad[0]);
    conv_layer<0><<<NATOM / 256, 256, 0, stream>>>(featB, wEbuf, Yv, nbr_idx, featA);
    // layer 1
    radial_mlp<<<EDGES / 128, 128, 0, stream>>>(basis, W1t[1], W2t[1], W3t[1],
                                                maskv, wEbuf, nwPad[1]);
    conv_layer<1><<<NATOM / 256, 256, 0, stream>>>(featA, wEbuf, Yv, nbr_idx, featB);
    // layer 2
    radial_mlp<<<EDGES / 128, 128, 0, stream>>>(basis, W1t[2], W2t[2], W3t[2],
                                                maskv, wEbuf, nwPad[2]);
    conv_layer<2><<<NATOM / 256, 256, 0, stream>>>(featB, wEbuf, Yv, nbr_idx, featA);
    // layer 3
    radial_mlp<<<EDGES / 128, 128, 0, stream>>>(basis, W1t[3], W2t[3], W3t[3],
                                                maskv, wEbuf, nwPad[3]);
    conv_layer<3><<<NATOM / 256, 256, 0, stream>>>(featA, wEbuf, Yv, nbr_idx, featB);

    // 4) pooling + linear head
    pool_kernel<<<64, 256, 0, stream>>>(featB, pooled);
    head_kernel<<<1, 32, 0, stream>>>(pooled, lin_w, lin_b, out);
}
